// DecoderLayer_86423331930134
// MI455X (gfx1250) — matrix operand order repaired and checked by reading, not compiled, not run on hardware
//
#include <hip/hip_runtime.h>
#include <hip/hip_bf16.h>

// ---------------------------------------------------------------------------
// GRU decoder (teacher forcing) for MI455X / gfx1250, wave32 + WMMA bf16.
//   B=32, T=128, H=1024, EMB=512, IN_DIM=1536, VOCAB=32000
// Pipeline:
//   1) build X = concat(emb[y], encoder_outputs) row-major bf16   [4096 x 1536]
//   2) pack W_in / U / Wo into WMMA B-fragment layout (bf16)
//   3) GEMM: mx = X @ W_in + b_in (async-LDS staged B, f32 accum) [4096 x 3072]
//   4) persistent GRU scan: 32 WGs x 4 waves; WG's U slice (192KB) pulled into
//      LDS once via TDM (tensor_load_to_lds); h tile in registers; WMMA from
//      LDS-resident B; atomic grid barrier per step.
//   5) GEMM: logits = hs @ Wo + bo, fused mask-zeroing -> d_out [B,T,V]
// ---------------------------------------------------------------------------

typedef __bf16 bf16x16 __attribute__((ext_vector_type(16)));
typedef __bf16 bf16x8  __attribute__((ext_vector_type(8)));
typedef float  f32x8   __attribute__((ext_vector_type(8)));
typedef unsigned int v4u __attribute__((ext_vector_type(4)));
typedef int v8i __attribute__((ext_vector_type(8)));
typedef int v4i __attribute__((ext_vector_type(4)));

#define B_    32
#define T_    128
#define H_    1024
#define EMB_  512
#define IND_  1536
#define V_    32000
#define M_    (T_ * B_)      /* 4096 */
#define N3H_  (3 * H_)       /* 3072 */
#define NWG_  32             /* workgroups in persistent scan */

#define SCHED_FENCE() __builtin_amdgcn_sched_barrier(0)

// Workspace layout (bytes)
static constexpr size_t OFF_XBF  = 0;                                   // 12.6 MB
static constexpr size_t OFF_WIN  = OFF_XBF  + (size_t)M_ * IND_ * 2;    //  9.4 MB
static constexpr size_t OFF_UPK  = OFF_WIN  + (size_t)IND_ * N3H_ * 2;  //  6.3 MB
static constexpr size_t OFF_WOPK = OFF_UPK  + (size_t)H_ * N3H_ * 2;    // 65.5 MB
static constexpr size_t OFF_MX   = OFF_WOPK + (size_t)H_ * V_ * 2;      // 50.3 MB
static constexpr size_t OFF_HBF  = OFF_MX   + (size_t)M_ * N3H_ * 4;    // 128 KB (x2 parity)
static constexpr size_t OFF_HSBF = OFF_HBF  + (size_t)2 * B_ * H_ * 2;  //  8.4 MB
static constexpr size_t OFF_CNT  = OFF_HSBF + (size_t)M_ * H_ * 2;      // barrier counter

// ---------------------------------------------------------------------------
// WMMA / fragment helpers
// ---------------------------------------------------------------------------
__device__ __forceinline__ f32x8 wmma_bf16(bf16x16 a, bf16x16 b, f32x8 c) {
  return __builtin_amdgcn_wmma_f32_16x16x32_bf16(false, a, false, b, (short)0, c,
                                                 false, false);
}

// A fragment (16x32 bf16, row-major source): lane l holds row m=(l&15),
// elems 0..7 = k{khalf*8+0..7}, elems 8..15 = k{16+khalf*8+0..7}, khalf=l>>4.
__device__ __forceinline__ bf16x16 load_a_frag(const __bf16* base) {
  bf16x8 lo = *(const bf16x8*)(base);
  bf16x8 hi = *(const bf16x8*)(base + 16);
  return __builtin_shufflevector(lo, hi, 0, 1, 2, 3, 4, 5, 6, 7,
                                 8, 9, 10, 11, 12, 13, 14, 15);
}

// Async global->LDS copy of 32 bytes (two b128 beats), tracked by ASYNCcnt.
__device__ __forceinline__ void async_copy32(uint32_t lds_addr, const void* gsrc) {
  asm volatile("global_load_async_to_lds_b128 %0, %1, off\n\t"
               "global_load_async_to_lds_b128 %0, %1, off offset:16"
               :: "v"(lds_addr), "v"(gsrc) : "memory");
}
__device__ __forceinline__ void wait_async0() {
  asm volatile("s_wait_asynccnt 0x0" ::: "memory");
}

// 1-D TDM transfer: n8 elements of 8 bytes, contiguous global -> LDS.
// D# group0: count=1, lds_addr, 57b global addr, type=2.
// D# group1: data_size=3 (8B), tensor_dim0=tile_dim0=n8 (1-D), stride=n8.
// clang-23 lane: 6-arg builtin (g0, g1, g2, g3, extra v8i, cpol).
__device__ __forceinline__ void tdm_load_1d(uint32_t lds_addr, const void* gsrc,
                                            uint32_t n8) {
  uint64_t ga = (uint64_t)(size_t)gsrc;
  v4u g0;
  g0[0] = 1u;                                                  // count=1
  g0[1] = lds_addr;                                            // lds_addr
  g0[2] = (uint32_t)ga;                                        // global_addr[31:0]
  g0[3] = (uint32_t)((ga >> 32) & 0x1FFFFFFu) | (2u << 30);    // addr[56:32], type=2
  v8i g1;
  g1[0] = 0x30000;                               // wg_mask=0, data_size=3 (8B)
  g1[1] = (int)((n8 & 0xFFFFu) << 16);           // tensor_dim0[15:0]
  g1[2] = (int)((n8 >> 16) & 0xFFFFu);           // tensor_dim0[31:16], dim1=0
  g1[3] = (int)((n8 & 0xFFFFu) << 16);           // tile_dim0 = n8
  g1[4] = 0;                                     // tile_dim1=0, tile_dim2=0 (1-D)
  g1[5] = (int)n8;                               // tensor_dim0_stride[31:0]
  g1[6] = 0;
  g1[7] = 0;
  v4i z4 = {0, 0, 0, 0};
  v8i z8 = {0, 0, 0, 0, 0, 0, 0, 0};
  __builtin_amdgcn_tensor_load_to_lds(g0, g1, z4, z4, z8, 0);
}

// ---------------------------------------------------------------------------
// init: h_bf[parity 0] = bf16(encoder_state), zero barrier counter
// ---------------------------------------------------------------------------
__global__ void __launch_bounds__(256) init_kernel(const float* __restrict__ es,
                                                   __bf16* __restrict__ hbf,
                                                   int* __restrict__ cnt) {
  int i = blockIdx.x * 256 + threadIdx.x;
  if (i < B_ * H_) hbf[i] = (__bf16)es[i];
  if (i == 0) *cnt = 0;
}

// ---------------------------------------------------------------------------
// build X (row-major bf16): X[m][k], m = t*32 + b
// ---------------------------------------------------------------------------
__global__ void __launch_bounds__(256) build_x_kernel(const int* __restrict__ y,
                                                      const float* __restrict__ emb,
                                                      const float* __restrict__ enc,
                                                      __bf16* __restrict__ Xbf) {
  int e = blockIdx.x * 256 + threadIdx.x;   // handles 4 consecutive k
  int m = e / (IND_ / 4);
  int k = (e % (IND_ / 4)) * 4;
  int t = m >> 5, b = m & 31;
  const float* src;
  if (k < EMB_) {
    int row = y[b * T_ + t];
    src = emb + (size_t)row * EMB_ + k;
  } else {
    src = enc + (size_t)b * H_ + (k - EMB_);
  }
  float4 v = *(const float4*)src;
  __bf16* dst = Xbf + (size_t)m * IND_ + k;
  dst[0] = (__bf16)v.x; dst[1] = (__bf16)v.y;
  dst[2] = (__bf16)v.z; dst[3] = (__bf16)v.w;
}

// ---------------------------------------------------------------------------
// pack B matrix [K x N] f32 row-major -> WMMA B fragments (bf16).
// ---------------------------------------------------------------------------
__global__ void __launch_bounds__(256) pack_b_kernel(const float* __restrict__ src,
                                                     __bf16* __restrict__ dst,
                                                     int K, int N) {
  int tid  = blockIdx.x * 256 + threadIdx.x;
  int lane = tid & 31;
  int tile = tid >> 5;
  int ktiles = K >> 5;
  int kt = tile % ktiles;
  int nt = tile / ktiles;
  if (nt >= (N >> 4)) return;
  int n = nt * 16 + (lane & 15);
  int kbase = kt * 32 + (lane >> 4) * 8;
  __bf16* d = dst + ((size_t)tile * 32 + lane) * 16;
#pragma unroll
  for (int e = 0; e < 16; ++e)
    d[e] = (__bf16)src[(size_t)(kbase + (e < 8 ? e : e + 8)) * N + n];
}

// ---------------------------------------------------------------------------
// GEMM: mx = Xbf[4096x1536] @ WinPack + b_in -> f32 [4096 x 3072]
// block 256 thr (8 waves); block tile 128M x 128N; wave 16M x 128N (8 C tiles).
// B panel (8KB/kt) double-buffered in LDS via async-to-LDS; B frags pipelined
// 1-deep with sched fences. kt loop kept rolled (unroll 1) so the accumulator
// phi coalesces with the in-place WMMA D==C (no v_mov ping-pong).
// ---------------------------------------------------------------------------
__global__ void __launch_bounds__(256) gemm_mx_kernel(const __bf16* __restrict__ Xbf,
                                                      const __bf16* __restrict__ Bpk,
                                                      const float* __restrict__ bin,
                                                      float* __restrict__ mx) {
  __shared__ char bsm[2][8192];
  const int tid = threadIdx.x;
  const int lane = tid & 31, w = tid >> 5;
  const int ln = lane & 15, khalf = lane >> 4;
  const int mbase = blockIdx.y * 128 + w * 16;
  const int ntG0 = blockIdx.x * 8;
  const int sc = tid >> 5;            // staging: tile 0..7
  const int soff = (tid & 31) * 32;   // staging: 32B chunk inside tile

  auto stage = [&](int kt, int buf) {
    const char* gsrc = (const char*)Bpk + ((size_t)(ntG0 + sc) * 48 + kt) * 1024 + soff;
    uint32_t laddr = (uint32_t)(size_t)&bsm[buf][sc * 1024 + soff];
    async_copy32(laddr, gsrc);
  };

  f32x8 acc[8] = {};
  stage(0, 0);
  wait_async0();
  __syncthreads();
  const __bf16* arow = Xbf + (size_t)(mbase + ln) * IND_ + khalf * 8;
  bf16x16 a_cur = load_a_frag(arow);
#pragma unroll 1
  for (int kt = 0; kt < 48; ++kt) {
    const int buf = kt & 1;
    const bool last = (kt == 47);
    if (!last) stage(kt + 1, buf ^ 1);
    bf16x16 a_nxt = a_cur;
    if (!last) a_nxt = load_a_frag(arow + (kt + 1) * 32);
    const char* bp = &bsm[buf][0];
    bf16x16 b_cur = *(const bf16x16*)(bp + lane * 32);
#pragma unroll
    for (int c = 0; c < 8; ++c) {
      bf16x16 b_nxt = b_cur;
      if (c < 7) b_nxt = *(const bf16x16*)(bp + (c + 1) * 1024 + lane * 32);
      acc[c] = wmma_bf16(a_cur, b_cur, acc[c]);
      SCHED_FENCE();
      b_cur = b_nxt;
    }
    a_cur = a_nxt;
    if (!last) { wait_async0(); __syncthreads(); }
  }
#pragma unroll
  for (int c = 0; c < 8; ++c) {
    int col = (ntG0 + c) * 16 + ln;
    float bv = bin[col];
#pragma unroll
    for (int j = 0; j < 8; ++j) {
      int m = mbase + j + 8 * khalf;           // C layout: row = j + 8*(lane>>4)
      mx[(size_t)m * N3H_ + col] = acc[c][j] + bv;
    }
  }
}

// ---------------------------------------------------------------------------
// Persistent GRU scan. 32 WGs x 128 thr (4 waves). WG wg owns h columns
// [wg*32, wg*32+32). Its U slice (3 gates x 2 ntiles x 32 ktiles = 192KB)
// is DMA'd into LDS once by the TDM and stays resident for all 128 steps.
// Per step: mx gate loads issued first (latency hidden behind 96 WMMAs/wave),
// recurrent GEMM from LDS-resident B with 1-deep pipelining, gate math in f32,
// bf16 h published to a double-buffered global array, atomic grid barrier.
// ---------------------------------------------------------------------------
__global__ void __launch_bounds__(128) gru_scan_kernel(const __bf16* __restrict__ Upk,
                                                       const float* __restrict__ mx,
                                                       const float* __restrict__ brec,
                                                       const float* __restrict__ es,
                                                       const int* __restrict__ mask,
                                                       __bf16* __restrict__ hbf,
                                                       __bf16* __restrict__ hsbf,
                                                       int* __restrict__ cnt) {
  extern __shared__ char smem[];                      // 6 panels x 32KB = 192KB
  const int wg = blockIdx.x;
  const int tid = threadIdx.x;
  const int lane = tid & 31, w = tid >> 5;
  const int ln = lane & 15, khalf = lane >> 4;
  const int mtile = w >> 1, ntq = w & 1;
  const int cl = wg * 32 + ntq * 16 + ln;             // owned h column

  // ---- TDM preload of U slice: panel pi = gate*2+q, ntG = gate*64 + wg*2 + q
  if (w == 0) {
    for (int pi = 0; pi < 6; ++pi) {
      int g = pi >> 1, q = pi & 1;
      int ntG = g * 64 + wg * 2 + q;
      tdm_load_1d((uint32_t)(size_t)(smem + (size_t)pi * 32768),
                  Upk + (size_t)ntG * 16384, 4096 /* 8B elements = 32KB */);
    }
    __builtin_amdgcn_s_wait_tensorcnt(0);
  }
  __syncthreads();

  const float bz = brec[cl], br = brec[H_ + cl], bh = brec[2 * H_ + cl];
  float hold[8];
  int mk[8];
#pragma unroll
  for (int j = 0; j < 8; ++j) {
    int b = mtile * 16 + j + 8 * khalf;
    hold[j] = es[(size_t)b * H_ + cl];
    mk[j] = mask[b];
  }
  const char* panz = smem + (size_t)(0 + ntq) * 32768;
  const char* panr = smem + (size_t)(2 + ntq) * 32768;
  const char* panh = smem + (size_t)(4 + ntq) * 32768;

  for (int t = 0; t < T_; ++t) {
    const int p = t & 1;
    // issue gate-input loads early; consumed after the GEMM
    float xz[8], xr[8], xh[8];
#pragma unroll
    for (int j = 0; j < 8; ++j) {
      const float* mb = mx + (size_t)(t * B_ + (mtile * 16 + j + 8 * khalf)) * N3H_ + cl;
      xz[j] = mb[0]; xr[j] = mb[H_]; xh[j] = mb[2 * H_];
    }
    f32x8 az = {}, ar = {}, ah = {};
    const __bf16* arow = hbf + (size_t)p * (B_ * H_) +
                         (size_t)(mtile * 16 + ln) * H_ + khalf * 8;
    bf16x16 a_cur = load_a_frag(arow);
    const int lb = lane * 32;
    bf16x16 b0 = *(const bf16x16*)(panz + lb);
    bf16x16 b1 = *(const bf16x16*)(panr + lb);
    bf16x16 b2 = *(const bf16x16*)(panh + lb);
#pragma unroll 1
    for (int kt = 0; kt < 32; ++kt) {
      const bool last = (kt == 31);
      bf16x16 a_nxt = a_cur, n0 = b0, n1 = b1, n2 = b2;
      if (!last) {
        a_nxt = load_a_frag(arow + (kt + 1) * 32);
        const int boff = (kt + 1) * 1024 + lb;
        n0 = *(const bf16x16*)(panz + boff);
        n1 = *(const bf16x16*)(panr + boff);
        n2 = *(const bf16x16*)(panh + boff);
      }
      az = wmma_bf16(a_cur, b0, az);
      ar = wmma_bf16(a_cur, b1, ar);
      ah = wmma_bf16(a_cur, b2, ah);
      SCHED_FENCE();
      a_cur = a_nxt; b0 = n0; b1 = n1; b2 = n2;
    }
#pragma unroll
    for (int j = 0; j < 8; ++j) {
      int b = mtile * 16 + j + 8 * khalf;
      size_t mrow = (size_t)(t * B_ + b);
      float z = 1.0f / (1.0f + __expf(-(xz[j] + az[j] + bz)));
      float r = 1.0f / (1.0f + __expf(-(xr[j] + ar[j] + br)));
      float hh = tanhf(xh[j] + r * (ah[j] + bh));
      float hn = z * hold[j] + (1.0f - z) * hh;
      hn = (t < mk[j]) ? hn : hold[j];                 // state passthrough
      hold[j] = hn;
      __bf16 hb = (__bf16)hn;
      hbf[(size_t)(1 - p) * (B_ * H_) + (size_t)b * H_ + cl] = hb;
      hsbf[mrow * H_ + cl] = hb;
    }
    // ---- grid barrier (monotonic counter; NWG_ resident WGs) ----
    __threadfence();
    __syncthreads();
    if (tid == 0) {
      __hip_atomic_fetch_add(cnt, 1, __ATOMIC_ACQ_REL, __HIP_MEMORY_SCOPE_AGENT);
      const int target = NWG_ * (t + 1);
      while (__hip_atomic_load(cnt, __ATOMIC_ACQUIRE, __HIP_MEMORY_SCOPE_AGENT) < target)
        __builtin_amdgcn_s_sleep(1);
    }
    __syncthreads();
  }
}

// ---------------------------------------------------------------------------
// GEMM: logits = hsbf[4096x1024] @ WoPack + bo, fused mask-zeroing.
// Same async-LDS staged + pipelined structure; grid (250, 32).
// ---------------------------------------------------------------------------
__global__ void __launch_bounds__(256) gemm_out_kernel(const __bf16* __restrict__ hsbf,
                                                       const __bf16* __restrict__ Bpk,
                                                       const float* __restrict__ bo,
                                                       const int* __restrict__ mask,
                                                       float* __restrict__ out) {
  __shared__ char bsm[2][8192];
  const int tid = threadIdx.x;
  const int lane = tid & 31, w = tid >> 5;
  const int ln = lane & 15, khalf = lane >> 4;
  const int mbase = blockIdx.y * 128 + w * 16;
  const int ntG0 = blockIdx.x * 8;
  const int sc = tid >> 5;
  const int soff = (tid & 31) * 32;

  auto stage = [&](int kt, int buf) {
    const char* gsrc = (const char*)Bpk + ((size_t)(ntG0 + sc) * 32 + kt) * 1024 + soff;
    uint32_t laddr = (uint32_t)(size_t)&bsm[buf][sc * 1024 + soff];
    async_copy32(laddr, gsrc);
  };

  f32x8 acc[8] = {};
  stage(0, 0);
  wait_async0();
  __syncthreads();
  const __bf16* arow = hsbf + (size_t)(mbase + ln) * H_ + khalf * 8;
  bf16x16 a_cur = load_a_frag(arow);
#pragma unroll 1
  for (int kt = 0; kt < 32; ++kt) {
    const int buf = kt & 1;
    const bool last = (kt == 31);
    if (!last) stage(kt + 1, buf ^ 1);
    bf16x16 a_nxt = a_cur;
    if (!last) a_nxt = load_a_frag(arow + (kt + 1) * 32);
    const char* bp = &bsm[buf][0];
    bf16x16 b_cur = *(const bf16x16*)(bp + lane * 32);
#pragma unroll
    for (int c = 0; c < 8; ++c) {
      bf16x16 b_nxt = b_cur;
      if (c < 7) b_nxt = *(const bf16x16*)(bp + (c + 1) * 1024 + lane * 32);
      acc[c] = wmma_bf16(a_cur, b_cur, acc[c]);
      SCHED_FENCE();
      b_cur = b_nxt;
    }
    a_cur = a_nxt;
    if (!last) { wait_async0(); __syncthreads(); }
  }
#pragma unroll
  for (int c = 0; c < 8; ++c) {
    int v = (ntG0 + c) * 16 + ln;
    float bv = bo[v];
#pragma unroll
    for (int j = 0; j < 8; ++j) {
      int m = mbase + j + 8 * khalf;
      int t = m >> 5, b = m & 31;
      float val = acc[c][j] + bv;
      out[((size_t)b * T_ + t) * V_ + v] = (t < mask[b]) ? val : 0.0f;
    }
  }
}

// ---------------------------------------------------------------------------
extern "C" void kernel_launch(void* const* d_in, const int* in_sizes, int n_in,
                              void* d_out, int out_size, void* d_ws, size_t ws_size,
                              hipStream_t stream) {
  const float* enc_out = (const float*)d_in[0];   // [B,H]
  const float* enc_st  = (const float*)d_in[1];   // [B,H]
  const int*   y       = (const int*)d_in[2];     // [B,T]
  const int*   mask    = (const int*)d_in[3];     // [B]
  const float* emb     = (const float*)d_in[4];   // [VOCAB,EMB]
  const float* W_in    = (const float*)d_in[5];   // [IND,3H]
  const float* b_in    = (const float*)d_in[6];   // [3H]
  const float* U       = (const float*)d_in[7];   // [H,3H]
  const float* b_rec   = (const float*)d_in[8];   // [3H]
  const float* Wo      = (const float*)d_in[9];   // [H,V]
  const float* bo      = (const float*)d_in[10];  // [V]
  float* out = (float*)d_out;

  char* ws = (char*)d_ws;
  __bf16* Xbf    = (__bf16*)(ws + OFF_XBF);
  __bf16* WinPk  = (__bf16*)(ws + OFF_WIN);
  __bf16* Upk    = (__bf16*)(ws + OFF_UPK);
  __bf16* WoPk   = (__bf16*)(ws + OFF_WOPK);
  float*  mx     = (float*)(ws + OFF_MX);
  __bf16* hbf    = (__bf16*)(ws + OFF_HBF);
  __bf16* hsbf   = (__bf16*)(ws + OFF_HSBF);
  int*    cnt    = (int*)(ws + OFF_CNT);

  // 1) init h state + barrier counter
  init_kernel<<<dim3((B_ * H_) / 256), dim3(256), 0, stream>>>(enc_st, hbf, cnt);
  // 2) build X (embedding gather + context concat, bf16)
  build_x_kernel<<<dim3((M_ * IND_ / 4) / 256), dim3(256), 0, stream>>>(y, emb, enc_out, Xbf);
  // 3) pack weights into WMMA B-fragment layout
  pack_b_kernel<<<dim3((IND_ / 32) * (N3H_ / 16) * 32 / 256), dim3(256), 0, stream>>>(W_in, WinPk, IND_, N3H_);
  pack_b_kernel<<<dim3((H_ / 32) * (N3H_ / 16) * 32 / 256), dim3(256), 0, stream>>>(U, Upk, H_, N3H_);
  pack_b_kernel<<<dim3((H_ / 32) * (V_ / 16) * 32 / 256), dim3(256), 0, stream>>>(Wo, WoPk, H_, V_);
  // 4) mx = X @ W_in + b_in
  gemm_mx_kernel<<<dim3(N3H_ / 128, M_ / 128), dim3(256), 0, stream>>>(Xbf, WinPk, b_in, mx);
  // 5) persistent GRU scan (192KB dynamic LDS: U slice resident per WG)
  gru_scan_kernel<<<dim3(NWG_), dim3(128), 196608, stream>>>(Upk, mx, b_rec, enc_st, mask, hbf, hsbf, cnt);
  // 6) logits = hs @ Wo + bo, masked, -> [B,T,V]
  gemm_out_kernel<<<dim3(V_ / 128, M_ / 128), dim3(256), 0, stream>>>(hsbf, WoPk, bo, mask, out);
  (void)in_sizes; (void)n_in; (void)out_size; (void)ws_size;
}
